// MultiHeadAttention_6133213299302
// MI455X (gfx1250) — matrix operand order repaired and checked by reading, not compiled, not run on hardware
//
#include <hip/hip_runtime.h>
#include <hip/hip_bf16.h>
#include <hip/hip_fp16.h>

// ---------------------------------------------------------------------------
// Relative-position MHA for gfx1250 (MI455X).
// wave32 + v_wmma_f32_16x16x32_f16 + LDS-fused attention +
// async global->LDS staging (ASYNCcnt) in the projection GEMMs.
// B=8, S=1024, D=1024, H=16, DH=64, MAX_K=4, SCALE=8.
// ---------------------------------------------------------------------------

typedef __attribute__((ext_vector_type(16))) _Float16 v16h;
typedef __attribute__((ext_vector_type(8)))  float    v8f;

#define B_  8
#define S_  1024
#define D_  1024
#define H_  16
#define DH_ 64
#define SCH 1040   // f16 score-tile row stride (1024 + 16 halfs pad)

// --- gfx1250 async global->LDS (tracked by ASYNCcnt) -----------------------
__device__ __forceinline__ void async_g2l_b128(unsigned lds_off,
                                               const void* gaddr) {
    asm volatile("global_load_async_to_lds_b128 %0, %1, off"
                 :: "v"(lds_off), "v"(gaddr)
                 : "memory");
}
__device__ __forceinline__ void wait_async0() {
    asm volatile("s_wait_asynccnt 0x0" ::: "memory");
}

// ---------------------------------------------------------------------------
// fp32 -> f16 conversion (one pass)
// ---------------------------------------------------------------------------
__global__ __launch_bounds__(256) void cvt_f16_kernel(
    const float* __restrict__ x, _Float16* __restrict__ y, int n)
{
    int i = (blockIdx.x * 256 + threadIdx.x) * 4;
    if (i + 3 < n) {
        float4 v = *(const float4*)(x + i);
        y[i]     = (_Float16)v.x;
        y[i + 1] = (_Float16)v.y;
        y[i + 2] = (_Float16)v.z;
        y[i + 3] = (_Float16)v.w;
    }
}

// ---------------------------------------------------------------------------
// Projection GEMM (f16 operands): Y[8192,1024] = X @ W^T + bias, K=1024.
// Double-buffered async global->LDS staging of 64x32 X and W slabs.
// MODE 0: f16 [B,H,S,DH] (Q,K)   MODE 1: f16 [B,H,DH,S] (V^T)
// MODE 2: fp32 row-major [M,N]   (final Wo projection -> d_out)
// 128 threads = 4 waves; WG tile 64x64; wave tile 32x32 (2x2 WMMA frags).
// ---------------------------------------------------------------------------
template <int MODE>
__global__ __launch_bounds__(128) void proj_gemm_f16_kernel(
    const _Float16* __restrict__ Xh, const _Float16* __restrict__ Wh,
    const float* __restrict__ bias, void* __restrict__ Y)
{
    __shared__ _Float16 sX[2][64][40];   // k-slab of X rows (stride 40 halfs)
    __shared__ _Float16 sW[2][64][40];   // k-slab of W rows

    const int K    = D_;
    const int tid  = threadIdx.x;
    const int lane = tid & 31;
    const int wave = tid >> 5;
    const int m16  = lane & 15;
    const int hi   = lane >> 4;
    const int tileMb = blockIdx.y * 64;
    const int tileNb = blockIdx.x * 64;
    const int mloc = (wave >> 1) * 32;
    const int nloc = (wave & 1) * 32;

    // per-thread staging source: row = tid>>2 (+0/+32), 8-half chunk = tid&3
    const int srow = tid >> 2, schunk = (tid & 3) * 8;
    const _Float16* gx0 = Xh + (size_t)(tileMb + srow) * K + schunk;
    const _Float16* gw0 = Wh + (size_t)(tileNb + srow) * K + schunk;

    auto stage = [&](int buf, int k0) {
        unsigned lx = (unsigned)(uintptr_t)&sX[buf][srow][schunk];
        unsigned lw = (unsigned)(uintptr_t)&sW[buf][srow][schunk];
        async_g2l_b128(lx,             gx0 + k0);
        async_g2l_b128(lx + 32 * 80,   gx0 + 32 * K + k0);   // rows 32..63
        async_g2l_b128(lw,             gw0 + k0);
        async_g2l_b128(lw + 32 * 80,   gw0 + 32 * K + k0);
    };

    v8f c[2][2] = {};

    stage(0, 0);
    wait_async0();
    __syncthreads();

    for (int ks = 0; ks < K / 32; ++ks) {
        const int cur = ks & 1;
        if (ks + 1 < K / 32) stage(cur ^ 1, (ks + 1) * 32);

        v16h a[2], bm[2];
#pragma unroll
        for (int mf = 0; mf < 2; ++mf) {
            const _Float16* xr = &sX[cur][mloc + mf * 16 + m16][0];
#pragma unroll
            for (int j = 0; j < 8; ++j) {
                a[mf][j]     = xr[16 * hi + j];
                a[mf][8 + j] = xr[16 * hi + 8 + j];
            }
        }
#pragma unroll
        for (int nf = 0; nf < 2; ++nf) {
            const _Float16* wr = &sW[cur][nloc + nf * 16 + m16][16 * hi];
#pragma unroll
            for (int j = 0; j < 16; ++j) bm[nf][j] = wr[j];
        }
#pragma unroll
        for (int mf = 0; mf < 2; ++mf)
#pragma unroll
            for (int nf = 0; nf < 2; ++nf)
                c[mf][nf] = __builtin_amdgcn_wmma_f32_16x16x32_f16(
                    false, a[mf], false, bm[nf], (short)0, c[mf][nf], false, false);

        if (ks + 1 < K / 32) {
            wait_async0();       // own transfers done ...
            __syncthreads();     // ... all waves' transfers visible
        }
    }

#pragma unroll
    for (int mf = 0; mf < 2; ++mf)
#pragma unroll
        for (int nf = 0; nf < 2; ++nf) {
            const int n  = tileNb + nloc + nf * 16 + m16;
            const float bv = bias[n];
#pragma unroll
            for (int r = 0; r < 8; ++r) {
                const int m = tileMb + mloc + mf * 16 + 8 * hi + r;
                const float val = c[mf][nf][r] + bv;
                if (MODE == 0) {
                    const int b = m >> 10, s = m & 1023, h = n >> 6, d = n & 63;
                    ((_Float16*)Y)[(((size_t)(b * H_ + h)) * S_ + s) * DH_ + d] =
                        (_Float16)val;
                } else if (MODE == 1) {
                    const int b = m >> 10, s = m & 1023, h = n >> 6, d = n & 63;
                    ((_Float16*)Y)[(((size_t)(b * H_ + h)) * DH_ + d) * S_ + s] =
                        (_Float16)val;
                } else {
                    ((float*)Y)[(size_t)m * D_ + n] = val;
                }
            }
        }
}

// ---------------------------------------------------------------------------
// Fused attention: one workgroup = (bh, 64 query rows), 512 threads = 16 waves.
// Score/attn tile lives in LDS as f16 (64 x 1024, stride SCH) -> ~150 KB LDS
// total, so TWO workgroups fit per 320 KB WGP.
//   phase 0: rdot[64][9] (q.pe_k) -> LDS; zero 64x64 out buffer
//   phase A: WMMA QK^T + rel bucket + mask, /8 -> f16 LDS tile
//   phase B: row softmax (fp32 math on f16 tile); ONE fp32 attn write to
//            d_out; normalized 9-bucket sums -> LDS
//   phase C: attn @ V via WMMA (A = f16 LDS tile, B = V^T f16 global),
//            split-K over 4 wave groups, ds_add_f32 into out buffer
//   phase D: += sum bs[d9]*pe_v[d9]; head output f16 [B*S, D]
// ---------------------------------------------------------------------------
__global__ __launch_bounds__(512) void fused_attn_kernel(
    const _Float16* __restrict__ Qh, const _Float16* __restrict__ Kh,
    const _Float16* __restrict__ Vt, const int* __restrict__ mask,
    const float* __restrict__ pe_k, const float* __restrict__ pe_v,
    float* __restrict__ attn, _Float16* __restrict__ headout)
{
    extern __shared__ char smemraw[];
    _Float16* scH = (_Float16*)smemraw;                      // [64][SCH] f16
    float* outb = (float*)(smemraw + 64 * SCH * 2);          // [64][64]
    float* rdv  = outb + 64 * 64;                            // [64][9]
    float* bsv  = rdv + 64 * 9;                              // [64][9]

    const int tid  = threadIdx.x;
    const int lane = tid & 31;
    const int wave = tid >> 5;
    const int m16  = lane & 15;
    const int hi   = lane >> 4;
    const int q0   = blockIdx.x * 64;
    const int bh   = blockIdx.y;
    const int b    = bh >> 4, h = bh & 15;
    const int* mrow = mask + b * S_;

    // ---- phase 0 ---------------------------------------------------------
    for (int i = tid; i < 64 * 64; i += 512) outb[i] = 0.f;
    for (int i = tid; i < 64 * 9; i += 512) {
        const int qloc = i / 9, d9 = i % 9;
        const _Float16* qr = Qh + ((size_t)bh * S_ + q0 + qloc) * DH_;
        const float*    p  = pe_k + d9 * DH_;
        float acc = 0.f;
#pragma unroll 8
        for (int j = 0; j < DH_; ++j) acc += (float)qr[j] * p[j];
        rdv[i] = acc;
    }
    __syncthreads();

    // ---- phase A: scores -> f16 LDS tile ---------------------------------
    {
        const int mloc = (wave >> 3) * 32;
        v16h a[2][2];                         // [kstep][mf]
#pragma unroll
        for (int ks = 0; ks < 2; ++ks)
#pragma unroll
            for (int mf = 0; mf < 2; ++mf) {
                const _Float16* qr =
                    Qh + ((size_t)bh * S_ + q0 + mloc + mf * 16 + m16) * DH_ +
                    ks * 32;
#pragma unroll
                for (int j = 0; j < 8; ++j) {
                    a[ks][mf][j]     = qr[16 * hi + j];
                    a[ks][mf][8 + j] = qr[16 * hi + 8 + j];
                }
            }
        for (int i = 0; i < 4; ++i) {
            const int n0 = ((wave & 7) * 4 + i) * 32;
            v16h bm[2][2];                    // [kstep][nf]
#pragma unroll
            for (int ks = 0; ks < 2; ++ks)
#pragma unroll
                for (int nf = 0; nf < 2; ++nf) {
                    const _Float16* kr =
                        Kh + ((size_t)bh * S_ + n0 + nf * 16 + m16) * DH_ +
                        ks * 32 + 16 * hi;
#pragma unroll
                    for (int j = 0; j < 16; ++j) bm[ks][nf][j] = kr[j];
                }
            v8f c[2][2] = {};
#pragma unroll
            for (int ks = 0; ks < 2; ++ks)
#pragma unroll
                for (int mf = 0; mf < 2; ++mf)
#pragma unroll
                    for (int nf = 0; nf < 2; ++nf)
                        c[mf][nf] = __builtin_amdgcn_wmma_f32_16x16x32_f16(
                            false, a[ks][mf], false, bm[ks][nf], (short)0,
                            c[mf][nf], false, false);
#pragma unroll
            for (int mf = 0; mf < 2; ++mf)
#pragma unroll
                for (int nf = 0; nf < 2; ++nf) {
                    const int kcol = n0 + nf * 16 + m16;
                    const bool masked = (mrow[kcol] == 0);
#pragma unroll
                    for (int r = 0; r < 8; ++r) {
                        const int qloc = mloc + mf * 16 + 8 * hi + r;
                        int dist = kcol - (q0 + qloc);
                        dist = dist < -4 ? -4 : (dist > 4 ? 4 : dist);
                        float s =
                            (c[mf][nf][r] + rdv[qloc * 9 + dist + 4]) * 0.125f;
                        if (masked) s = -__builtin_inff();
                        scH[qloc * SCH + kcol] = (_Float16)s;
                    }
                }
        }
    }
    __syncthreads();

    // ---- phase B: softmax + attn write + bucket sums ---------------------
    {
        float* attn_g = attn + (size_t)bh * S_ * S_;
#pragma unroll
        for (int rr = 0; rr < 4; ++rr) {
            const int qloc = wave * 4 + rr;
            _Float16* rowp = scH + qloc * SCH;
            const int qpos = q0 + qloc;

            float mx = -__builtin_inff();
            for (int j = lane; j < S_; j += 32) mx = fmaxf(mx, (float)rowp[j]);
#pragma unroll
            for (int off = 16; off > 0; off >>= 1)
                mx = fmaxf(mx, __shfl_xor(mx, off, 32));

            float sum = 0.f, bs[9];
#pragma unroll
            for (int d = 0; d < 9; ++d) bs[d] = 0.f;
            for (int j = lane; j < S_; j += 32) {
                const float e = __expf((float)rowp[j] - mx);  // masked -> 0
                rowp[j] = (_Float16)e;
                sum += e;
                int dist = j - qpos;
                dist = dist < -4 ? -4 : (dist > 4 ? 4 : dist);
                bs[dist + 4] += e;
            }
#pragma unroll
            for (int off = 16; off > 0; off >>= 1) sum += __shfl_xor(sum, off, 32);
            const float inv = 1.0f / sum;

            for (int j = lane; j < S_; j += 32) {
                const float an = (float)rowp[j] * inv;
                rowp[j] = (_Float16)an;                     // feeds phase C
                attn_g[(size_t)qpos * S_ + j] = an;         // mandated fp32 out
            }
#pragma unroll
            for (int d = 0; d < 9; ++d) {
                float v = bs[d];
#pragma unroll
                for (int off = 16; off > 0; off >>= 1)
                    v += __shfl_xor(v, off, 32);
                if (lane == 0) bsv[qloc * 9 + d] = v * inv;
            }
        }
    }
    __syncthreads();

    // ---- phase C: attn @ V (A from f16 LDS tile, split-K x4) -------------
    {
        const int mh = wave & 1, nh = (wave >> 1) & 1, kq = wave >> 2;
        v8f c[2][2] = {};
        const int kend = kq * 256 + 256;
        for (int k0 = kq * 256; k0 < kend; k0 += 32) {
            v16h a[2], bm[2];
#pragma unroll
            for (int mf = 0; mf < 2; ++mf) {
                const _Float16* ar = scH + (mh * 32 + mf * 16 + m16) * SCH + k0;
#pragma unroll
                for (int j = 0; j < 8; ++j) {
                    a[mf][j]     = ar[16 * hi + j];
                    a[mf][8 + j] = ar[16 * hi + 8 + j];
                }
            }
#pragma unroll
            for (int nf = 0; nf < 2; ++nf) {
                const _Float16* vr =
                    Vt + ((size_t)bh * DH_ + nh * 32 + nf * 16 + m16) * S_ +
                    k0 + 16 * hi;
#pragma unroll
                for (int j = 0; j < 16; ++j) bm[nf][j] = vr[j];
            }
#pragma unroll
            for (int mf = 0; mf < 2; ++mf)
#pragma unroll
                for (int nf = 0; nf < 2; ++nf)
                    c[mf][nf] = __builtin_amdgcn_wmma_f32_16x16x32_f16(
                        false, a[mf], false, bm[nf], (short)0, c[mf][nf], false,
                        false);
        }
#pragma unroll
        for (int mf = 0; mf < 2; ++mf)
#pragma unroll
            for (int nf = 0; nf < 2; ++nf) {
                const int d = nh * 32 + nf * 16 + m16;
#pragma unroll
                for (int r = 0; r < 8; ++r) {
                    const int qloc = mh * 32 + mf * 16 + 8 * hi + r;
                    atomicAdd(&outb[qloc * 64 + d], c[mf][nf][r]);
                }
            }
    }
    __syncthreads();

    // ---- phase D: + rel_v, write head output (f16, [B*S, D]) ------------
    for (int i = tid; i < 64 * 64; i += 512) {
        const int row = i >> 6, d = i & 63;
        float val = outb[i];
        const float* bsr = bsv + row * 9;
#pragma unroll
        for (int t = 0; t < 9; ++t) val += bsr[t] * pe_v[t * DH_ + d];
        headout[((size_t)b * S_ + q0 + row) * D_ + h * DH_ + d] = (_Float16)val;
    }
}

// ---------------------------------------------------------------------------
extern "C" void kernel_launch(void* const* d_in, const int* in_sizes, int n_in,
                              void* d_out, int out_size, void* d_ws, size_t ws_size,
                              hipStream_t stream)
{
    const float* query = (const float*)d_in[0];
    const float* key   = (const float*)d_in[1];
    const float* value = (const float*)d_in[2];
    const int*   mask  = (const int*)  d_in[3];
    const float* Wq = (const float*)d_in[4];
    const float* bq = (const float*)d_in[5];
    const float* Wk = (const float*)d_in[6];
    const float* bk = (const float*)d_in[7];
    const float* Wv = (const float*)d_in[8];
    const float* bv = (const float*)d_in[9];
    const float* Wo = (const float*)d_in[10];
    const float* bo = (const float*)d_in[11];
    const float* pe_k = (const float*)d_in[12];
    const float* pe_v = (const float*)d_in[13];

    float* out  = (float*)d_out;                              // [B,S,D]
    float* attn = (float*)d_out + (size_t)B_ * S_ * D_;       // [B,H,S,S]

    char* ws = (char*)d_ws;
    const size_t MB = 1024ull * 1024ull;
    _Float16* Qh  = (_Float16*)(ws);                          // 16 MB
    _Float16* Kh  = (_Float16*)(ws + 16 * MB);                // 16 MB
    _Float16* Vth = (_Float16*)(ws + 32 * MB);                // 16 MB (V^T)
    _Float16* Xh  = (_Float16*)(ws + 48 * MB);                // 16 MB (reused)
    _Float16* Wh  = (_Float16*)(ws + 64 * MB);                // 4 x 2 MB
    _Float16* Wh0 = Wh;
    _Float16* Wh1 = Wh + 1 * 1048576;
    _Float16* Wh2 = Wh + 2 * 1048576;
    _Float16* Wh3 = Wh + 3 * 1048576;

    const int nW = D_ * D_;            // 1,048,576
    const int nX = B_ * S_ * D_;       // 8,388,608
    const dim3 cblkW((nW / 4 + 255) / 256), cblkX((nX / 4 + 255) / 256);
    cvt_f16_kernel<<<cblkW, 256, 0, stream>>>(Wq, Wh0, nW);
    cvt_f16_kernel<<<cblkW, 256, 0, stream>>>(Wk, Wh1, nW);
    cvt_f16_kernel<<<cblkW, 256, 0, stream>>>(Wv, Wh2, nW);
    cvt_f16_kernel<<<cblkW, 256, 0, stream>>>(Wo, Wh3, nW);

    const dim3 blk(128);
    const dim3 gproj(D_ / 64, (B_ * S_) / 64);                // (16, 128)

    cvt_f16_kernel<<<cblkX, 256, 0, stream>>>(query, Xh, nX);
    proj_gemm_f16_kernel<0><<<gproj, blk, 0, stream>>>(Xh, Wh0, bq, (void*)Qh);
    cvt_f16_kernel<<<cblkX, 256, 0, stream>>>(key, Xh, nX);
    proj_gemm_f16_kernel<0><<<gproj, blk, 0, stream>>>(Xh, Wh1, bk, (void*)Kh);
    cvt_f16_kernel<<<cblkX, 256, 0, stream>>>(value, Xh, nX);
    proj_gemm_f16_kernel<1><<<gproj, blk, 0, stream>>>(Xh, Wh2, bv, (void*)Vth);

    // Fused scores+softmax+attn@V; writes attn (fp32) and head output (f16->Xh)
    const size_t lds_bytes =
        (size_t)64 * SCH * 2 + (64 * 64 + 64 * 9 + 64 * 9) * sizeof(float);
    const dim3 gfa(S_ / 64, B_ * H_);                         // (16, 128)
    fused_attn_kernel<<<gfa, 512, lds_bytes, stream>>>(
        Qh, Kh, Vth, mask, pe_k, pe_v, attn, Xh);

    proj_gemm_f16_kernel<2><<<gproj, blk, 0, stream>>>(Xh, Wh3, bo, (void*)out);
}
